// BertGCN_86139864089311
// MI455X (gfx1250) — matrix operand order repaired and checked by reading, not compiled, not run on hardware
//
#include <hip/hip_runtime.h>
#include <math.h>

// ---- problem constants (from reference) ----
#define BB   32
#define LL   512
#define NN   (BB*LL)        // 16384 nodes
#define HID  768
#define VSZ  8192
#define LAB  10

// ---- WMMA types (CDNA5 gfx1250, wave32) ----
typedef __attribute__((ext_vector_type(16))) __bf16 v16bf;
typedef __attribute__((ext_vector_type(8)))  float  v8f;

static __device__ __forceinline__ v8f wmma_bf16(v16bf a, v16bf b, v8f c) {
  // D = A(16x32 bf16) * B(32x16 bf16) + C(16x16 f32)
  return __builtin_amdgcn_wmma_f32_16x16x32_bf16(false, a, false, b, (short)0, c, false, false);
}

// A fragment from an f32 row (hardware v_cvt_pk_bf16_f32). 16-bit A 16x32 layout:
//  element e (0..15): k = ((e&8)<<1) + (laneHalf<<3) + (e&7)
static __device__ __forceinline__ v16bf load_afrag_f32(const float* __restrict__ arow, int kbase, int half) {
  v16bf u;
#pragma unroll
  for (int e8 = 0; e8 < 8; ++e8) {
    int kp = ((e8 & 4) << 2) + (half << 3) + ((e8 & 3) << 1); // pair base k
    float2 f = *(const float2*)(arow + kbase + kp);
    u[2*e8]   = (__bf16)f.x;
    u[2*e8+1] = (__bf16)f.y;
  }
  return u;
}

// Fragment-major packed loads: each lane's 16 bf16 elements are contiguous (32B).
static __device__ __forceinline__ v16bf load_frag_packed(const __bf16* __restrict__ P, size_t fragIdx, int lane) {
  const uint4* p = (const uint4*)(P + (fragIdx * 32 + lane) * 16);
  union { v16bf v; uint4 q[2]; } u;
  u.q[0] = p[0];
  u.q[1] = p[1];
  return u.v;
}

// A-fragment packed index for element (row n, k) of an [*,32] activation:
//  mt = n/16, lane = (n&15) | ((k>>3 & 1)<<4), e = ((k>>4)<<3) | (k&7)
static __device__ __forceinline__ size_t a_pack_idx(int n, int k) {
  int mt = n >> 4;
  int l  = (n & 15) | (((k >> 3) & 1) << 4);
  int e  = (((k >> 4) & 1) << 3) | (k & 7);
  return ((size_t)mt * 32 + l) * 16 + e;
}

static __device__ __forceinline__ float lrelu(float x) { return x > 0.f ? x : 0.2f * x; }

// ---------------- K0: pack f32 weight [K x C] into bf16 WMMA-B fragment layout ----------------
// B layout per 32x16 tile: lane = (c&15) | ((k>>3 & 1)<<4), e = ((k>>4 & 1)<<3) | (k&7)
__global__ void k_pack_b(const float* __restrict__ X, __bf16* __restrict__ Y, int K, int C) {
  int i = blockIdx.x * blockDim.x + threadIdx.x;
  if (i >= K * C) return;
  int k = i / C, c = i % C;
  int kt = k >> 5, ct = c >> 4;
  int l  = (c & 15) | (((k >> 3) & 1) << 4);
  int e  = (((k >> 4) & 1) << 3) | (k & 7);
  size_t po = (((size_t)kt * (C >> 4) + ct) * 32 + l) * 16 + e;
  Y[po] = (__bf16)X[i];
}

// ---------------- K1: h1 = features @ W1  ([N,768]x[768,32]) ----------------
__global__ void k_gemm1(const float* __restrict__ X, const __bf16* __restrict__ Wp,
                        float* __restrict__ H) {
  int wave = threadIdx.x >> 5, lane = threadIdx.x & 31;
  int tileM = blockIdx.x * (blockDim.x >> 5) + wave;
  int half = lane >> 4;
  int row  = tileM * 16 + (lane & 15);
  const float* arow = X + (size_t)row * HID;
  int col0 = lane & 15, col1 = 16 + (lane & 15);
  v8f c0 = {}; v8f c1 = {};
  for (int kt = 0; kt < HID / 32; ++kt) {
    __builtin_prefetch(arow + kt * 32 + 128, 0, 0);      // global_prefetch_b8
    v16bf a  = load_afrag_f32(arow, kt * 32, half);
    v16bf b0 = load_frag_packed(Wp, (size_t)kt * 2 + 0, lane);
    v16bf b1 = load_frag_packed(Wp, (size_t)kt * 2 + 1, lane);
    c0 = wmma_bf16(a, b0, c0);
    c1 = wmma_bf16(a, b1, c1);
  }
#pragma unroll
  for (int r = 0; r < 8; ++r) {
    int ro = tileM * 16 + r + half * 8;                  // f32 C/D layout
    H[(size_t)ro * 32 + col0] = c0[r];
    H[(size_t)ro * 32 + col1] = c1[r];
  }
}

// ---------------- K1b: attention logits for conv1 (per node, 2 heads) ----------------
__global__ void k_att1(const float* __restrict__ H, const float* __restrict__ as1,
                       const float* __restrict__ ad1, float* __restrict__ aS, float* __restrict__ aD) {
  int i = blockIdx.x * blockDim.x + threadIdx.x;    // n*2 + head
  if (i >= NN * 2) return;
  int n = i >> 1, h = i & 1;
  const float* hp = H + (size_t)n * 32 + h * 16;
  float s = 0.f, d = 0.f;
#pragma unroll
  for (int c = 0; c < 16; ++c) { s += hp[c] * as1[h*16+c]; d += hp[c] * ad1[h*16+c]; }
  aS[i] = s; aD[i] = d;
}

// ---------------- K2: conv1 stencil (both branches) + bias + relu -> bf16 A-fragment layout ----------------
__global__ void k_conv1(const float* __restrict__ H, const float* __restrict__ aS,
                        const float* __restrict__ aD, const float* __restrict__ b1,
                        const int* __restrict__ tokS, const int* __restrict__ tokO,
                        const float* __restrict__ EM,
                        __bf16* __restrict__ x1s, __bf16* __restrict__ x1o) {
  int i = blockIdx.x * blockDim.x + threadIdx.x;
  if (i >= NN * 32) return;
  int n = i >> 5, col = i & 31, head = col >> 4;
  int p = n & (LL - 1);
  bool hasL = (p > 0), hasR = (p < LL - 1);
  float adn = aD[n*2 + head];
  float eL = hasL ? lrelu(aS[(n-1)*2 + head] + adn) : -1e30f;
  float eR = hasR ? lrelu(aS[(n+1)*2 + head] + adn) : -1e30f;
  float eS = lrelu(aS[n*2 + head] + adn);
  float m  = fmaxf(eS, fmaxf(eL, eR));
  float xL = hasL ? expf(eL - m) : 0.f;
  float xR = hasR ? expf(eR - m) : 0.f;
  float xS = expf(eS - m);
  float inv = 1.f / (xL + xR + xS + 1e-16f);
  float aL = xL * inv, aR = xR * inv, aSf = xS * inv;
  float hL = hasL ? H[(size_t)(n-1)*32 + col] : 0.f;
  float hR = hasR ? H[(size_t)(n+1)*32 + col] : 0.f;
  float hS = H[(size_t)n*32 + col];
  float bc = b1[col];
  size_t po = a_pack_idx(n, col);
  // branch 0 (src tokens)
  {
    int tn = tokS[n];
    float ewL = hasL ? EM[(size_t)tokS[n-1]*VSZ + tn] : 0.f;
    float ewR = hasR ? EM[(size_t)tokS[n+1]*VSZ + tn] : 0.f;
    float o = aL*ewL*hL + aR*ewR*hR + aSf*hS + bc;
    x1s[po] = (__bf16)fmaxf(o, 0.f);
  }
  // branch 1 (other tokens)
  {
    int tn = tokO[n];
    float ewL = hasL ? EM[(size_t)tokO[n-1]*VSZ + tn] : 0.f;
    float ewR = hasR ? EM[(size_t)tokO[n+1]*VSZ + tn] : 0.f;
    float o = aL*ewL*hL + aR*ewR*hR + aSf*hS + bc;
    x1o[po] = (__bf16)fmaxf(o, 0.f);
  }
}

// ---------------- K3: h2 = x1 @ W2  ([N,32]x[32,768]) — single WMMA per tile, all-packed operands ----------------
__global__ void k_gemm2(const __bf16* __restrict__ X1p, const __bf16* __restrict__ W2p,
                        float* __restrict__ H2) {
  int wave = threadIdx.x >> 5, lane = threadIdx.x & 31;
  int tileM = blockIdx.y * (blockDim.x >> 5) + wave;
  int tileN = blockIdx.x;
  int half = lane >> 4;
  int col = tileN * 16 + (lane & 15);
  v16bf a = load_frag_packed(X1p, (size_t)tileM, lane);
  v16bf b = load_frag_packed(W2p, (size_t)tileN, lane);   // K=32 -> single K-tile
  v8f c = {};
  c = wmma_bf16(a, b, c);
#pragma unroll
  for (int r = 0; r < 8; ++r) {
    int ro = tileM * 16 + r + half * 8;
    H2[(size_t)ro * HID + col] = c[r];
  }
}

// ---------------- K4: conv2 attention dots (wave32 per node, shuffle reduce) ----------------
__global__ void k_att2(const float* __restrict__ H2, const float* __restrict__ as2,
                       const float* __restrict__ ad2, float* __restrict__ aS2, float* __restrict__ aD2) {
  int wave = threadIdx.x >> 5, lane = threadIdx.x & 31;
  int n = blockIdx.x * (blockDim.x >> 5) + wave;
  const float* hp = H2 + (size_t)n * HID;
  float s = 0.f, d = 0.f;
  for (int c = lane; c < HID; c += 32) { float h = hp[c]; s += h * as2[c]; d += h * ad2[c]; }
#pragma unroll
  for (int off = 16; off > 0; off >>= 1) { s += __shfl_down(s, off, 32); d += __shfl_down(d, off, 32); }
  if (lane == 0) { aS2[n] = s; aD2[n] = d; }
}

// ---------------- K5: conv2 stencil aggregation fused with mean pooling ----------------
__global__ void k_conv2_pool(const float* __restrict__ H2, const float* __restrict__ aS2,
                             const float* __restrict__ aD2, const float* __restrict__ b2,
                             float* __restrict__ pooled) {
  int i = blockIdx.x * blockDim.x + threadIdx.x;
  if (i >= BB * HID) return;
  int bb = i / HID, c = i % HID;
  float acc = 0.f;
  for (int p = 0; p < LL; ++p) {
    int n = bb * LL + p;
    bool hasL = (p > 0), hasR = (p < LL - 1);
    float adn = aD2[n];
    float eL = hasL ? lrelu(aS2[n-1] + adn) : -1e30f;
    float eR = hasR ? lrelu(aS2[n+1] + adn) : -1e30f;
    float eS = lrelu(aS2[n] + adn);
    float m  = fmaxf(eS, fmaxf(eL, eR));
    float xL = hasL ? expf(eL - m) : 0.f;
    float xR = hasR ? expf(eR - m) : 0.f;
    float xS = expf(eS - m);
    float inv = 1.f / (xL + xR + xS + 1e-16f);
    float s = xS * H2[(size_t)n * HID + c];
    if (hasL) s += xL * H2[(size_t)(n-1) * HID + c];
    if (hasR) s += xR * H2[(size_t)(n+1) * HID + c];
    acc += s * inv;
  }
  pooled[i] = acc / (float)LL + b2[c];
}

// ---------------- K6: o1 = tanh(pooled @ Wo1 + bo1)  ([32,768]x[768,768]) ----------------
__global__ void k_out1(const float* __restrict__ P, const __bf16* __restrict__ Wo1p,
                       const float* __restrict__ bo1, float* __restrict__ O1) {
  int lane = threadIdx.x & 31;
  int tileM = blockIdx.y, tileN = blockIdx.x;
  int half = lane >> 4;
  int row = tileM * 16 + (lane & 15);
  int col = tileN * 16 + (lane & 15);
  const float* arow = P + (size_t)row * HID;
  v8f c = {};
  for (int kt = 0; kt < HID / 32; ++kt) {
    __builtin_prefetch(arow + kt * 32 + 128, 0, 0);
    v16bf a = load_afrag_f32(arow, kt * 32, half);
    v16bf b = load_frag_packed(Wo1p, (size_t)kt * (HID / 16) + tileN, lane);
    c = wmma_bf16(a, b, c);
  }
#pragma unroll
  for (int r = 0; r < 8; ++r) {
    int ro = tileM * 16 + r + half * 8;
    O1[(size_t)ro * HID + col] = tanhf(c[r] + bo1[col]);
  }
}

// ---------------- K7: logits = o1 @ Wo2 + bo2 ----------------
__global__ void k_logits(const float* __restrict__ O1, const float* __restrict__ Wo2,
                         const float* __restrict__ bo2, float* __restrict__ Lg) {
  int i = blockIdx.x * blockDim.x + threadIdx.x;
  if (i >= BB * LAB) return;
  int r = i / LAB, j = i % LAB;
  float s = bo2[j];
  for (int c = 0; c < HID; ++c) s += O1[(size_t)r * HID + c] * Wo2[(size_t)c * LAB + j];
  Lg[i] = s;
}

// ---------------- K8a: SimCSE row normalization (interleaved out/o_out) ----------------
__global__ void k_norm(const float* __restrict__ O1s, const float* __restrict__ O1o,
                       float* __restrict__ PN) {
  int r = blockIdx.x; // 0..63 ; even -> branch0 row r/2, odd -> branch1
  const float* src = (r & 1) ? (O1o + (size_t)(r >> 1) * HID) : (O1s + (size_t)(r >> 1) * HID);
  __shared__ float red[256];
  float s = 0.f;
  for (int c = threadIdx.x; c < HID; c += blockDim.x) { float v = src[c]; s += v * v; }
  red[threadIdx.x] = s; __syncthreads();
  for (int st = 128; st > 0; st >>= 1) { if (threadIdx.x < st) red[threadIdx.x] += red[threadIdx.x + st]; __syncthreads(); }
  float inv = 1.f / fmaxf(sqrtf(red[0]), 1e-8f);
  for (int c = threadIdx.x; c < HID; c += blockDim.x) PN[(size_t)r * HID + c] = src[c] * inv;
}

// ---------------- K8b: sim = (pn pn^T - 1e12*I) / 0.05 ----------------
__global__ void k_sim(const float* __restrict__ PN, float* __restrict__ S) {
  int i = blockIdx.x * blockDim.x + threadIdx.x;
  if (i >= 64 * 64) return;
  int r = i >> 6, cj = i & 63;
  float s = 0.f;
  for (int c = 0; c < HID; ++c) s += PN[(size_t)r * HID + c] * PN[(size_t)cj * HID + c];
  if (r == cj) s -= 1e12f;
  S[i] = s * 20.f; // / 0.05
}

// ---------------- K8c: total_loss = nll(logits) + nll(o_logits) + simcse ----------------
__global__ void k_loss(const float* __restrict__ S, const float* __restrict__ Lg0,
                       const float* __restrict__ Lg1, const int* __restrict__ label,
                       float* __restrict__ out) {
  __shared__ float red[64];
  int t = threadIdx.x; // 64 threads
  float m = -1e30f;
  for (int j = 0; j < 64; ++j) m = fmaxf(m, S[t*64 + j]);
  float z = 0.f;
  for (int j = 0; j < 64; ++j) z += expf(S[t*64 + j] - m);
  float rl = logf(z) + m - S[t*64 + (t ^ 1)];   // partner = t^1
  red[t] = rl; __syncthreads();
  for (int st = 32; st > 0; st >>= 1) { if (t < st) red[t] += red[t + st]; __syncthreads(); }
  float sloss = red[0] / 64.f;
  __syncthreads();
  float nl = 0.f;
  if (t < BB) {
    int lab = label[t];
    for (int br = 0; br < 2; ++br) {
      const float* row = (br ? Lg1 : Lg0) + t * LAB;
      float mm = row[0];
      for (int j = 1; j < LAB; ++j) mm = fmaxf(mm, row[j]);
      float zz = 0.f;
      for (int j = 0; j < LAB; ++j) zz += expf(row[j] - mm);
      nl += logf(zz) + mm - row[lab];
    }
  }
  red[t] = (t < BB) ? nl : 0.f; __syncthreads();
  for (int st = 32; st > 0; st >>= 1) { if (t < st) red[t] += red[t + st]; __syncthreads(); }
  if (t == 0) out[0] = red[0] / (float)BB + sloss;
}

// ---------------- host launch ----------------
extern "C" void kernel_launch(void* const* d_in, const int* in_sizes, int n_in,
                              void* d_out, int out_size, void* d_ws, size_t ws_size,
                              hipStream_t stream) {
  const int*   tokS = (const int*)  d_in[0];   // src [B,L]
  const int*   tokO = (const int*)  d_in[1];   // other_input_ids [B,L]
  const int*   lab  = (const int*)  d_in[2];   // label [B]
  const float* feat = (const float*)d_in[3];   // [N,768]
  const float* EM   = (const float*)d_in[4];   // [V,V]
  const float* W1   = (const float*)d_in[5];   // [768,32]
  const float* as1  = (const float*)d_in[6];   // [2,16]
  const float* ad1  = (const float*)d_in[7];
  const float* b1   = (const float*)d_in[8];   // [32]
  const float* W2   = (const float*)d_in[9];   // [32,768]
  const float* as2  = (const float*)d_in[10];  // [768]
  const float* ad2  = (const float*)d_in[11];
  const float* b2   = (const float*)d_in[12];  // [768]
  const float* Wo1  = (const float*)d_in[13];  // [768,768]
  const float* bo1  = (const float*)d_in[14];
  const float* Wo2  = (const float*)d_in[15];  // [768,10]
  const float* bo2  = (const float*)d_in[16];

  char* base = (char*)d_ws;
  size_t off = 0;
  auto alloc = [&](size_t bytes) -> char* {
    char* p = base + off; off += (bytes + 255) & ~(size_t)255; return p;
  };
  float*  h1    = (float*) alloc((size_t)NN * 32 * 4);
  float*  aS1   = (float*) alloc((size_t)NN * 2 * 4);
  float*  aD1   = (float*) alloc((size_t)NN * 2 * 4);
  __bf16* x1s   = (__bf16*)alloc((size_t)NN * 32 * 2);    // A-fragment packed
  __bf16* x1o   = (__bf16*)alloc((size_t)NN * 32 * 2);    // A-fragment packed
  float*  h2    = (float*) alloc((size_t)NN * HID * 4);   // reused across branches
  float*  aS2   = (float*) alloc((size_t)NN * 4);
  float*  aD2   = (float*) alloc((size_t)NN * 4);
  float*  pool0 = (float*) alloc((size_t)BB * HID * 4);
  float*  pool1 = (float*) alloc((size_t)BB * HID * 4);
  float*  o1_0  = (float*) alloc((size_t)BB * HID * 4);
  float*  o1_1  = (float*) alloc((size_t)BB * HID * 4);
  float*  lgO   = (float*) alloc((size_t)BB * LAB * 4);
  float*  pn    = (float*) alloc((size_t)64 * HID * 4);
  float*  sim   = (float*) alloc((size_t)64 * 64 * 4);
  __bf16* wb1   = (__bf16*)alloc((size_t)HID * 32 * 2);   // B-fragment packed
  __bf16* wb2   = (__bf16*)alloc((size_t)32 * HID * 2);   // B-fragment packed
  __bf16* wbo1  = (__bf16*)alloc((size_t)HID * HID * 2);  // B-fragment packed

  float* outp = (float*)d_out;       // [0]=total_loss, [1..320]=logits (branch 0)
  float* lgS  = outp + 1;

  // one-shot weight packing (f32 -> bf16 fragment-major)
  k_pack_b<<<(HID*32  + 255)/256, 256, 0, stream>>>(W1,  wb1,  HID, 32);
  k_pack_b<<<(32*HID  + 255)/256, 256, 0, stream>>>(W2,  wb2,  32,  HID);
  k_pack_b<<<(HID*HID + 255)/256, 256, 0, stream>>>(Wo1, wbo1, HID, HID);

  // shared stage
  k_gemm1<<<NN/16/4, 128, 0, stream>>>(feat, wb1, h1);
  k_att1 <<<(NN*2 + 255)/256, 256, 0, stream>>>(h1, as1, ad1, aS1, aD1);
  k_conv1<<<(NN*32 + 255)/256, 256, 0, stream>>>(h1, aS1, aD1, b1, tokS, tokO, EM, x1s, x1o);

  // per-branch pipeline (sequential on stream; h2/aS2/aD2 reused)
  for (int br = 0; br < 2; ++br) {
    const __bf16* x1 = br ? x1o : x1s;
    float* pooled    = br ? pool1 : pool0;
    float* o1        = br ? o1_1 : o1_0;
    float* lg        = br ? lgO  : lgS;
    k_gemm2<<<dim3(HID/16, NN/16/4), 128, 0, stream>>>(x1, wb2, h2);
    k_att2 <<<NN/8, 256, 0, stream>>>(h2, as2, ad2, aS2, aD2);
    k_conv2_pool<<<(BB*HID + 255)/256, 256, 0, stream>>>(h2, aS2, aD2, b2, pooled);
    k_out1 <<<dim3(HID/16, BB/16), 32, 0, stream>>>(pooled, wbo1, bo1, o1);
    k_logits<<<2, 160, 0, stream>>>(o1, Wo2, bo2, lg);
  }

  // losses
  k_norm<<<64, 256, 0, stream>>>(o1_0, o1_1, pn);
  k_sim <<<16, 256, 0, stream>>>(pn, sim);
  k_loss<<<1, 64, 0, stream>>>(sim, lgS, lgO, lab, outp);
}
